// AttentionHead_45629732553164
// MI455X (gfx1250) — matrix operand order repaired and checked by reading, not compiled, not run on hardware
//
#include <hip/hip_runtime.h>

#define BATCH 16
#define SEQ   2048
#define EMB   1024
#define HD    64

typedef __attribute__((ext_vector_type(16))) __bf16 v16bf;
typedef __attribute__((ext_vector_type(8)))  __bf16 v8bf;
typedef __attribute__((ext_vector_type(8)))  float  v8f;
typedef __attribute__((ext_vector_type(4)))  float  v4f;

static __device__ __forceinline__ v8f wmma_bf16(v16bf a, v16bf b, v8f c) {
  // D = A(16x32 bf16) x B(32x16 bf16) + C(16x16 f32)
  return __builtin_amdgcn_wmma_f32_16x16x32_bf16(false, a, false, b, (short)0, c,
                                                 false, false);
}

static __device__ __forceinline__ v8f v8f_zero() {
  v8f z = {0.f, 0.f, 0.f, 0.f, 0.f, 0.f, 0.f, 0.f};
  return z;
}

// ---------------------------------------------------------------------------
// Kernel 1: Wk/Wq/Wv [E][H] fp32  ->  Wt [H][E] bf16 (B-operand friendly)
// ---------------------------------------------------------------------------
__global__ void wconv_kernel(const float* __restrict__ Wk,
                             const float* __restrict__ Wq,
                             const float* __restrict__ Wv,
                             unsigned short* __restrict__ Wkt_u,
                             unsigned short* __restrict__ Wqt_u,
                             unsigned short* __restrict__ Wvt_u) {
  __bf16* Wkt = (__bf16*)Wkt_u;
  __bf16* Wqt = (__bf16*)Wqt_u;
  __bf16* Wvt = (__bf16*)Wvt_u;
  int idx = blockIdx.x * blockDim.x + threadIdx.x;
  if (idx >= EMB * HD) return;
  int e = idx / HD, h = idx % HD;
  size_t t = (size_t)h * EMB + e;
  Wkt[t] = (__bf16)Wk[idx];
  Wqt[t] = (__bf16)Wq[idx];
  Wvt[t] = (__bf16)Wv[idx];
}

// ---------------------------------------------------------------------------
// Kernel 2: projections.  One wave = 16 rows of t, full K loop over E.
//   Q,K stored row-major bf16 [B][T][H];  V stored transposed bf16 [B][H][T].
// ---------------------------------------------------------------------------
__global__ __launch_bounds__(256) void proj_kernel(
    const float* __restrict__ X,
    const unsigned short* __restrict__ Wqt_u,
    const unsigned short* __restrict__ Wkt_u,
    const unsigned short* __restrict__ Wvt_u,
    unsigned short* __restrict__ Qu,
    unsigned short* __restrict__ Ku,
    unsigned short* __restrict__ Vtu) {
  const __bf16* Wqt = (const __bf16*)Wqt_u;
  const __bf16* Wkt = (const __bf16*)Wkt_u;
  const __bf16* Wvt = (const __bf16*)Wvt_u;
  __bf16* Q  = (__bf16*)Qu;
  __bf16* K  = (__bf16*)Ku;
  __bf16* Vt = (__bf16*)Vtu;

  const int wave = threadIdx.x >> 5;
  const int lane = threadIdx.x & 31;
  const int col  = lane & 15;
  const int half = lane >> 4;
  const int b    = blockIdx.x >> 4;   // SEQ/128 == 16 t-blocks per batch
  const int tblk = blockIdx.x & 15;
  const int t0   = tblk * 128 + wave * 16;

  const float* xrow = X + ((size_t)b * SEQ + t0 + col) * EMB;

  v8f accQ[4], accK[4], accV[4];
#pragma unroll
  for (int i = 0; i < 4; i++) { accQ[i] = v8f_zero(); accK[i] = v8f_zero(); accV[i] = v8f_zero(); }

  for (int e0 = 0; e0 < EMB; e0 += 32) {
    // A operand: 16x32 bf16 tile of X, contiguous k = 16*half + s layout.
    v4f f0 = *(const v4f*)(xrow + e0 + half * 16);
    v4f f1 = *(const v4f*)(xrow + e0 + half * 16 + 4);
    v4f f2 = *(const v4f*)(xrow + e0 + half * 16 + 8);
    v4f f3 = *(const v4f*)(xrow + e0 + half * 16 + 12);
    v16bf a;
#pragma unroll
    for (int i = 0; i < 4; i++) {
      a[i]      = (__bf16)f0[i];
      a[4 + i]  = (__bf16)f1[i];
      a[8 + i]  = (__bf16)f2[i];
      a[12 + i] = (__bf16)f3[i];
    }
#pragma unroll
    for (int ht = 0; ht < 4; ht++) {
      size_t wrow = (size_t)(ht * 16 + col) * EMB + e0 + half * 16;
      v16bf bq = *(const v16bf*)(Wqt + wrow);
      v16bf bk = *(const v16bf*)(Wkt + wrow);
      v16bf bv = *(const v16bf*)(Wvt + wrow);
      accQ[ht] = wmma_bf16(a, bq, accQ[ht]);
      accK[ht] = wmma_bf16(a, bk, accK[ht]);
      accV[ht] = wmma_bf16(a, bv, accV[ht]);
    }
  }

#pragma unroll
  for (int ht = 0; ht < 4; ht++) {
#pragma unroll
    for (int r = 0; r < 8; r++) {
      int t = t0 + r + half * 8;                  // D layout: M = r + 8*half
      size_t o = ((size_t)b * SEQ + t) * HD + ht * 16 + col;
      Q[o] = (__bf16)accQ[ht][r];
      K[o] = (__bf16)accK[ht][r];
    }
    v8bf pv;                                      // V transposed: contiguous store
#pragma unroll
    for (int r = 0; r < 8; r++) pv[r] = (__bf16)accV[ht][r];
    *(v8bf*)(Vt + ((size_t)b * HD + ht * 16 + col) * SEQ + t0 + half * 8) = pv;
  }
}

// ---------------------------------------------------------------------------
// Kernel 3: causal flash attention.  One wave = 16 queries; 32-key blocks.
//   S^T = K·Q^T  ->  per-lane online softmax (1 shfl per reduction)
//   O^T += V^T · P^T  (P converted D-layout -> B-layout via wave-private LDS)
// ---------------------------------------------------------------------------
__global__ __launch_bounds__(256) void attn_kernel(
    const unsigned short* __restrict__ Qu,
    const unsigned short* __restrict__ Ku,
    const unsigned short* __restrict__ Vtu,
    float* __restrict__ Out) {
  __shared__ __bf16 ldsP[8][16][32];              // per-wave 16x32 P tile (1KB)

  const __bf16* Q  = (const __bf16*)Qu;
  const __bf16* K  = (const __bf16*)Ku;
  const __bf16* Vt = (const __bf16*)Vtu;

  const int wave = threadIdx.x >> 5;
  const int lane = threadIdx.x & 31;
  const int col  = lane & 15;
  const int half = lane >> 4;
  const int b    = blockIdx.x >> 4;
  const int tblk = blockIdx.x & 15;
  const int q0   = tblk * 128 + wave * 16;
  const int myq  = q0 + col;                      // this lane's query row
  const float sc = 0.125f * 1.4426950408889634f;  // 1/sqrt(H) * log2(e)

  // Q as B operand (column n of Q^T == row n of Q): contiguous 16-elem chunks
  const __bf16* qrow = Q + ((size_t)b * SEQ + q0 + col) * HD;
  v16bf Bq0 = *(const v16bf*)(qrow + half * 16);
  v16bf Bq1 = *(const v16bf*)(qrow + 32 + half * 16);

  v8f Oacc[4];
#pragma unroll
  for (int i = 0; i < 4; i++) Oacc[i] = v8f_zero();
  float m = -1e30f, l = 0.f;

  const int kkend = ((q0 + 16 + 31) >> 5) << 5;   // last (diagonal) block incl.

  for (int kk = 0; kk < kkend; kk += 32) {
    // ---- S^T tiles: A = K rows (keys), B = Q rows -------------------------
    v8f S[2];
#pragma unroll
    for (int j = 0; j < 2; j++) {
      const __bf16* krow = K + ((size_t)b * SEQ + kk + j * 16 + col) * HD;
      v8bf x0 = *(const v8bf*)(krow + half * 16);
      v8bf x1 = *(const v8bf*)(krow + half * 16 + 8);
      v8bf x2 = *(const v8bf*)(krow + 32 + half * 16);
      v8bf x3 = *(const v8bf*)(krow + 40 + half * 16);
      v16bf ka0, ka1;
#pragma unroll
      for (int i = 0; i < 8; i++) {
        ka0[i] = x0[i]; ka0[8 + i] = x1[i];
        ka1[i] = x2[i]; ka1[8 + i] = x3[i];
      }
      v8f s = v8f_zero();
      s = wmma_bf16(ka0, Bq0, s);
      s = wmma_bf16(ka1, Bq1, s);
      S[j] = s;
    }

    // ---- scale + causal mask + online softmax (per-lane: one query) -------
    float ps[16];
    float mloc = -1e30f;
#pragma unroll
    for (int j = 0; j < 2; j++) {
#pragma unroll
      for (int r = 0; r < 8; r++) {
        int key = kk + j * 16 + half * 8 + r;     // S^T: M = key index
        float v = S[j][r] * sc;
        v = (key <= myq) ? v : -1e30f;
        ps[j * 8 + r] = v;
        mloc = fmaxf(mloc, v);
      }
    }
    mloc = fmaxf(mloc, __shfl_xor(mloc, 16, 32)); // combine key halves
    float m_new = fmaxf(m, mloc);
    float corr = exp2f(m - m_new);
    m = m_new;

    float sl = 0.f;
    v8bf p0, p1;
#pragma unroll
    for (int r = 0; r < 8; r++) {
      float e0v = exp2f(ps[r] - m_new);
      float e1v = exp2f(ps[8 + r] - m_new);
      sl += e0v + e1v;
      p0[r] = (__bf16)e0v;
      p1[r] = (__bf16)e1v;
    }
    sl += __shfl_xor(sl, 16, 32);
    l = l * corr + sl;
#pragma unroll
    for (int ht = 0; ht < 4; ht++)
#pragma unroll
      for (int r = 0; r < 8; r++) Oacc[ht][r] *= corr;

    // ---- P: D-layout -> LDS [query][key] -> B operand ---------------------
    *(v8bf*)&ldsP[wave][col][half * 8]      = p0;  // keys kk+0..15
    *(v8bf*)&ldsP[wave][col][16 + half * 8] = p1;  // keys kk+16..31
    asm volatile("s_wait_dscnt 0" ::: "memory");   // wave-private, in-order LDS
    v16bf Bp = *(const v16bf*)&ldsP[wave][col][half * 16];

    // ---- O^T += V^T(16hd x 32key) · P^T(32key x 16q) ----------------------
#pragma unroll
    for (int ht = 0; ht < 4; ht++) {
      const __bf16* vrow = Vt + ((size_t)b * HD + ht * 16 + col) * SEQ + kk;
      v8bf y0 = *(const v8bf*)(vrow + half * 16);
      v8bf y1 = *(const v8bf*)(vrow + half * 16 + 8);
      v16bf va;
#pragma unroll
      for (int i = 0; i < 8; i++) { va[i] = y0[i]; va[8 + i] = y1[i]; }
      Oacc[ht] = wmma_bf16(va, Bp, Oacc[ht]);
    }
  }

  // ---- epilogue: normalize, store O (O^T D-layout => contiguous per lane) --
  float inv = 1.0f / l;
  float* orow = Out + ((size_t)b * SEQ + q0 + col) * HD;
#pragma unroll
  for (int ht = 0; ht < 4; ht++) {
    v4f o0, o1;
#pragma unroll
    for (int r = 0; r < 4; r++) {
      o0[r] = Oacc[ht][r] * inv;
      o1[r] = Oacc[ht][4 + r] * inv;
    }
    *(v4f*)(orow + ht * 16 + half * 8)     = o0;
    *(v4f*)(orow + ht * 16 + half * 8 + 4) = o1;
  }
}

// ---------------------------------------------------------------------------
extern "C" void kernel_launch(void* const* d_in, const int* in_sizes, int n_in,
                              void* d_out, int out_size, void* d_ws, size_t ws_size,
                              hipStream_t stream) {
  const float* X  = (const float*)d_in[0];  // embeddings [B,T,E]
  const float* Wk = (const float*)d_in[1];  // [E,H]
  const float* Wq = (const float*)d_in[2];
  const float* Wv = (const float*)d_in[3];
  float* out = (float*)d_out;               // [B,T,H] fp32

  size_t off = 0;
  char* base = (char*)d_ws;
  auto carve = [&](size_t bytes) -> unsigned short* {
    unsigned short* p = (unsigned short*)(base + off);
    off += (bytes + 255) & ~(size_t)255;
    return p;
  };
  unsigned short* Wkt = carve((size_t)EMB * HD * 2);
  unsigned short* Wqt = carve((size_t)EMB * HD * 2);
  unsigned short* Wvt = carve((size_t)EMB * HD * 2);
  unsigned short* Qb  = carve((size_t)BATCH * SEQ * HD * 2);
  unsigned short* Kb  = carve((size_t)BATCH * SEQ * HD * 2);
  unsigned short* Vtb = carve((size_t)BATCH * SEQ * HD * 2);

  wconv_kernel<<<(EMB * HD + 255) / 256, 256, 0, stream>>>(Wk, Wq, Wv, Wkt, Wqt, Wvt);
  proj_kernel<<<BATCH * (SEQ / 128), 256, 0, stream>>>(X, Wqt, Wkt, Wvt, Qb, Kb, Vtb);
  attn_kernel<<<BATCH * (SEQ / 128), 256, 0, stream>>>(Qb, Kb, Vtb, out);
}
